// MultiHeadAttention_79087527788784
// MI455X (gfx1250) — compile-verified, matrix operand order checked by reading, not run on hardware
//
#include <hip/hip_runtime.h>
#include <hip/hip_bf16.h>

// MHA block for MI455X (gfx1250): bf16 WMMA everywhere, f32 softmax/LN.
// B=8, L=1024, D=768, H=12, DK=DV=64, M=B*L=8192.
// Attention is computed fully transposed (S^T, O^T) so the exp(P) values
// stay in the WMMA B-fragment register layout (no LDS staging) and the
// softmax stats are one scalar per lane (one shfl_xor(16) per reduction).
// Both GEMM and attention use register-prefetch double buffering so next-tile
// global loads overlap the WMMA + VALU work of the current tile.

#define DEV static __device__ __forceinline__

typedef __attribute__((ext_vector_type(16))) __bf16 v16bf;
typedef __attribute__((ext_vector_type(8)))  __bf16 v8bf;
typedef __attribute__((ext_vector_type(8)))  float  v8f;

DEV unsigned short f2bf(float f) {
  unsigned int u = __float_as_uint(f);
  u += 0x7FFFu + ((u >> 16) & 1u);       // round-to-nearest-even
  return (unsigned short)(u >> 16);
}

DEV v8f wmma_bf16(v16bf a, v16bf b, v8f c) {
  // v_wmma_f32_16x16x32_bf16: D = A(16x32) * B(32x16) + C
  return __builtin_amdgcn_wmma_f32_16x16x32_bf16(false, a, false, b, (short)0, c,
                                                 false, false);
}

// Load a 16x32 A- or B-fragment from LDS stored row-major [16][ld] bf16.
// Per ISA: lanes 0-15 hold row=lane, K 0..7 & 16..23; lanes 16-31 hold
// row=lane-16, K 8..15 & 24..31.  Two 16-byte ds_load_b128 per lane.
DEV v16bf load_frag(const unsigned short* base, int ld, int lane) {
  int row = lane & 15;
  int off = (lane >> 4) * 8;
  const __bf16* p = reinterpret_cast<const __bf16*>(base) + row * ld + off;
  v8bf a0 = *reinterpret_cast<const v8bf*>(p);        // K off..off+7
  v8bf a1 = *reinterpret_cast<const v8bf*>(p + 16);   // K off+16..off+23
  return __builtin_shufflevector(a0, a1, 0,1,2,3,4,5,6,7,8,9,10,11,12,13,14,15);
}

// log-depth in-lane reductions over 16 values
DEV float max16(const float* f) {
  float t[8];
#pragma unroll
  for (int j = 0; j < 8; ++j) t[j] = fmaxf(f[j], f[j + 8]);
#pragma unroll
  for (int s = 4; s > 0; s >>= 1)
#pragma unroll
    for (int j = 0; j < 4; ++j) if (j < s) t[j] = fmaxf(t[j], t[j + s]);
  return t[0];
}
DEV float sum16(const float* f) {
  float t[8];
#pragma unroll
  for (int j = 0; j < 8; ++j) t[j] = f[j] + f[j + 8];
#pragma unroll
  for (int s = 4; s > 0; s >>= 1)
#pragma unroll
    for (int j = 0; j < 4; ++j) if (j < s) t[j] = t[j] + t[j + s];
  return t[0];
}

// ---------------------------------------------------------------- stage 0
__global__ __launch_bounds__(256)
void cvt_bf16_kernel(const float* __restrict__ in, unsigned short* __restrict__ out, int n) {
  int i = (blockIdx.x * 256 + threadIdx.x) * 4;
  if (i + 3 < n) {
    float4 f = *reinterpret_cast<const float4*>(in + i);
    ushort4 r;
    r.x = f2bf(f.x); r.y = f2bf(f.y); r.z = f2bf(f.z); r.w = f2bf(f.w);
    *reinterpret_cast<ushort4*>(out + i) = r;
  }
}

// W[768 k][768 n] f32 -> WT[768 n][768 k] bf16, both sides coalesced via
// a 32x32 LDS tile (stride-768 f32 reads would be ~8x amplified otherwise).
__global__ __launch_bounds__(256)
void wtrans_bf16_kernel(const float* __restrict__ W, unsigned short* __restrict__ WT) {
  __shared__ unsigned short tile[32][33];
  const int bk = blockIdx.x * 32;           // input row (k) base
  const int bn = blockIdx.y * 32;           // input col (n) base
  const int tx = threadIdx.x & 31, ty = threadIdx.x >> 5;
#pragma unroll
  for (int i = 0; i < 32; i += 8)
    tile[ty + i][tx] = f2bf(W[(size_t)(bk + ty + i) * 768 + bn + tx]);
  __syncthreads();
#pragma unroll
  for (int i = 0; i < 32; i += 8)
    WT[(size_t)(bn + ty + i) * 768 + bk + tx] = tile[tx][ty + i];
}

// ---------------------------------------------------------------- GEMM
// C[M=8192, N=768] = A[M,768] * BT^T  (BT stored [N][K] bf16), + bias.
// MODE 0: out bf16, head-major [B][H][L][64], value scaled by `scale`.
// MODE 1: out f32 row-major + residual (f32) add.
template <int MODE>
__global__ __launch_bounds__(256)
void gemm_bf16_kernel(const unsigned short* __restrict__ A,
                      const unsigned short* __restrict__ BT,
                      const float* __restrict__ bias,
                      const float* __restrict__ residual,
                      void* __restrict__ Cout, float scale) {
  __shared__ __align__(16) unsigned short As[128][32];
  __shared__ __align__(16) unsigned short Bs[128][32];

  const int bm = blockIdx.x * 128;
  const int bn = blockIdx.y * 128;
  const int tid = threadIdx.x, lane = tid & 31, wid = tid >> 5;
  const int wm = (wid >> 2) * 64;   // wave row offset in tile
  const int wn = (wid & 3) * 32;    // wave col offset in tile

  v8f acc[4][2];
#pragma unroll
  for (int it = 0; it < 4; ++it)
#pragma unroll
    for (int jt = 0; jt < 2; ++jt)
#pragma unroll
      for (int e = 0; e < 8; ++e) acc[it][jt][e] = 0.f;

  const int r = tid >> 2;
  const int cc = (tid & 3) * 8;
  const unsigned short* Ap0 = A + (size_t)(bm + r) * 768 + cc;
  const unsigned short* Ap1 = A + (size_t)(bm + r + 64) * 768 + cc;
  const unsigned short* Bp0 = BT + (size_t)(bn + r) * 768 + cc;
  const unsigned short* Bp1 = BT + (size_t)(bn + r + 64) * 768 + cc;

  uint4 pa[2], pb[2];
  pa[0] = *reinterpret_cast<const uint4*>(Ap0);
  pa[1] = *reinterpret_cast<const uint4*>(Ap1);
  pb[0] = *reinterpret_cast<const uint4*>(Bp0);
  pb[1] = *reinterpret_cast<const uint4*>(Bp1);

  for (int k0 = 0; k0 < 768; k0 += 32) {
    __syncthreads();
    *reinterpret_cast<uint4*>(&As[r][cc])      = pa[0];
    *reinterpret_cast<uint4*>(&As[r + 64][cc]) = pa[1];
    *reinterpret_cast<uint4*>(&Bs[r][cc])      = pb[0];
    *reinterpret_cast<uint4*>(&Bs[r + 64][cc]) = pb[1];
    __syncthreads();

    if (k0 + 32 < 768) {  // prefetch next K-slice while WMMAs run
      pa[0] = *reinterpret_cast<const uint4*>(Ap0 + k0 + 32);
      pa[1] = *reinterpret_cast<const uint4*>(Ap1 + k0 + 32);
      pb[0] = *reinterpret_cast<const uint4*>(Bp0 + k0 + 32);
      pb[1] = *reinterpret_cast<const uint4*>(Bp1 + k0 + 32);
    }

    v16bf af[4], bfr[2];
#pragma unroll
    for (int it = 0; it < 4; ++it) af[it] = load_frag(&As[wm + it * 16][0], 32, lane);
#pragma unroll
    for (int jt = 0; jt < 2; ++jt) bfr[jt] = load_frag(&Bs[wn + jt * 16][0], 32, lane);
#pragma unroll
    for (int it = 0; it < 4; ++it)
#pragma unroll
      for (int jt = 0; jt < 2; ++jt) acc[it][jt] = wmma_bf16(af[it], bfr[jt], acc[it][jt]);
  }

  const int hi = lane >> 4, ln = lane & 15;
#pragma unroll
  for (int it = 0; it < 4; ++it)
#pragma unroll
    for (int jt = 0; jt < 2; ++jt)
#pragma unroll
      for (int p = 0; p < 8; ++p) {
        int row = bm + wm + it * 16 + p + 8 * hi;
        int col = bn + wn + jt * 16 + ln;
        float v = (acc[it][jt][p] + bias[col]) * scale;
        if (MODE == 0) {
          int b = row >> 10, l = row & 1023, h = col >> 6, dk = col & 63;
          ((unsigned short*)Cout)[(((size_t)b * 12 + h) * 1024 + l) * 64 + dk] = f2bf(v);
        } else {
          size_t o = (size_t)row * 768 + col;
          ((float*)Cout)[o] = v + residual[o];
        }
      }
}

// ---------------------------------------------------------------- attention
// One block = (b, h, 64 q-rows).  4 waves, each wave 16 q-cols x all 64 dv.
// Fully transposed: S^T = K * Q^T, O^T = V^T * P^T.  Each lane owns one q
// column: softmax stats are per-lane scalars, P^T stays in registers in the
// exact B-fragment layout required by the AV WMMA.
__global__ __launch_bounds__(128)
void attn_kernel(const unsigned short* __restrict__ Qh,
                 const unsigned short* __restrict__ Kh,
                 const unsigned short* __restrict__ Vh,
                 const unsigned char* __restrict__ mask,
                 unsigned short* __restrict__ attB) {
  const int qt = blockIdx.x, h = blockIdx.y, b = blockIdx.z;
  const int tid = threadIdx.x, lane = tid & 31, wid = tid >> 5;
  const int hi = lane >> 4, ln = lane & 15;
  const int qb0 = qt * 64;
  const size_t headoff = ((size_t)b * 12 + h) * 1024 * 64;
  const unsigned short* Qp = Qh + headoff;

  __shared__ __align__(16) unsigned short Qs[64][64];
  __shared__ __align__(16) unsigned short Ks[32][64];   // [key][dk]
  __shared__ __align__(16) unsigned short VsT[64][32];  // [dv][key]
  __shared__ unsigned int mbits[32];                    // 1024-key padding bitmask

  { // Q tile 64x64
    int rr = tid >> 1, c = (tid & 1) * 32;
#pragma unroll
    for (int i = 0; i < 4; ++i)
      *reinterpret_cast<uint4*>(&Qs[rr][c + i * 8]) =
          *reinterpret_cast<const uint4*>(&Qp[(size_t)(qb0 + rr) * 64 + c + i * 8]);
  }
  if (tid < 32) { // pack padding mask into bits
    const unsigned char* mp = mask + (size_t)b * 1024 + tid * 32;
    unsigned w = 0;
#pragma unroll
    for (int j = 0; j < 32; ++j) w |= (mp[j] ? 1u : 0u) << j;
    mbits[tid] = w;
  }

  // per-thread K/V tile slice pointers + first-tile register prefetch
  const int key = tid & 31, c = (tid >> 5) * 16;
  const unsigned short* Kbase = Kh + headoff + (size_t)key * 64 + c;
  const unsigned short* Vbase = Vh + headoff + (size_t)key * 64 + c;
  uint4 rk0 = *reinterpret_cast<const uint4*>(Kbase);
  uint4 rk1 = *reinterpret_cast<const uint4*>(Kbase + 8);
  uint4 rv0 = *reinterpret_cast<const uint4*>(Vbase);
  uint4 rv1 = *reinterpret_cast<const uint4*>(Vbase + 8);

  __syncthreads();

  // Q^T as B-fragments (lane = q column), loaded once
  v16bf qf0 = load_frag(&Qs[wid * 16][0], 64, lane);   // dk 0..31
  v16bf qf1 = load_frag(&Qs[wid * 16][32], 64, lane);  // dk 32..63

  v8f accF[4], accB[4];
#pragma unroll
  for (int t = 0; t < 4; ++t)
#pragma unroll
    for (int e = 0; e < 8; ++e) { accF[t][e] = 0.f; accB[t][e] = 0.f; }
  float mF = -1e30f, lF = 0.f, mB = -1e30f, lB = 0.f;

  const int q = qb0 + wid * 16 + ln;   // this lane's q row
  const float NEG = -1e9f;

  for (int kk = 0; kk < 1024; kk += 32) {
    __syncthreads();
    { // commit prefetched K tile [key][dk]; V tile transposed -> VsT[dv][key]
      *reinterpret_cast<uint4*>(&Ks[key][c])     = rk0;
      *reinterpret_cast<uint4*>(&Ks[key][c + 8]) = rk1;
      union { uint4 u; unsigned short s[8]; } v0, v1;
      v0.u = rv0; v1.u = rv1;
#pragma unroll
      for (int j = 0; j < 8; ++j) {
        VsT[c + j][key] = v0.s[j];
        VsT[c + 8 + j][key] = v1.s[j];
      }
    }
    __syncthreads();

    if (kk + 32 < 1024) {  // prefetch next tile under compute
      const unsigned short* kp = Kbase + (size_t)(kk + 32) * 64;
      const unsigned short* vp = Vbase + (size_t)(kk + 32) * 64;
      rk0 = *reinterpret_cast<const uint4*>(kp);
      rk1 = *reinterpret_cast<const uint4*>(kp + 8);
      rv0 = *reinterpret_cast<const uint4*>(vp);
      rv1 = *reinterpret_cast<const uint4*>(vp + 8);
    }

    // S^T tiles: rows = keys (VGPR idx), cols = q (lanes).  dk = 2 chunks.
    v8f st0, st1;
#pragma unroll
    for (int e = 0; e < 8; ++e) { st0[e] = 0.f; st1[e] = 0.f; }
    st0 = wmma_bf16(load_frag(&Ks[0][0], 64, lane), qf0, st0);
    st0 = wmma_bf16(load_frag(&Ks[0][32], 64, lane), qf1, st0);
    st1 = wmma_bf16(load_frag(&Ks[16][0], 64, lane), qf0, st1);
    st1 = wmma_bf16(load_frag(&Ks[16][32], 64, lane), qf1, st1);

    const unsigned mw = mbits[kk >> 5];
    float f[16], g[16];
#pragma unroll
    for (int p = 0; p < 8; ++p) {
      int key0 = kk + p + 8 * hi;       // S^T row -> global key index
      int key1 = key0 + 16;
      float sc0 = ((mw >> (p + 8 * hi)) & 1u) ? NEG : st0[p];
      float sc1 = ((mw >> (16 + p + 8 * hi)) & 1u) ? NEG : st1[p];
      f[p]     = sc0 + ((key0 < q) ? NEG : 0.f);   // fw: attend k >= q
      f[8 + p] = sc1 + ((key1 < q) ? NEG : 0.f);
      g[p]     = sc0 + ((key0 > q) ? NEG : 0.f);   // bw: attend k <= q
      g[8 + p] = sc1 + ((key1 > q) ? NEG : 0.f);
    }

    // --- fw online softmax (per-lane stats; one xor-16 pairs the halves)
    float mxF = max16(f);
    mxF = fmaxf(mxF, __shfl_xor(mxF, 16, 32));
    float mnF = fmaxf(mF, mxF);
    float cF = __expf(mF - mnF);
    mF = mnF;
    v16bf pF;
    float eF[16];
#pragma unroll
    for (int j = 0; j < 16; ++j) { eF[j] = __expf(f[j] - mnF); pF[j] = (__bf16)eF[j]; }
    float sF = sum16(eF);
    sF += __shfl_xor(sF, 16, 32);
    lF = lF * cF + sF;

    // --- bw online softmax
    float mxB = max16(g);
    mxB = fmaxf(mxB, __shfl_xor(mxB, 16, 32));
    float mnB = fmaxf(mB, mxB);
    float cB = __expf(mB - mnB);
    mB = mnB;
    v16bf pB;
    float eB[16];
#pragma unroll
    for (int j = 0; j < 16; ++j) { eB[j] = __expf(g[j] - mnB); pB[j] = (__bf16)eB[j]; }
    float sB = sum16(eB);
    sB += __shfl_xor(sB, 16, 32);
    lB = lB * cB + sB;

    // rescale accumulators (scalar per lane) then O^T += V^T * P^T
#pragma unroll
    for (int t = 0; t < 4; ++t)
#pragma unroll
      for (int e = 0; e < 8; ++e) { accF[t][e] *= cF; accB[t][e] *= cB; }
#pragma unroll
    for (int t = 0; t < 4; ++t) {
      v16bf vf = load_frag(&VsT[t * 16][0], 32, lane);  // A: 16 dv x 32 keys
      accF[t] = wmma_bf16(vf, pF, accF[t]);
      accB[t] = wmma_bf16(vf, pB, accB[t]);
    }
  }

  // epilogue: O = accF/lF + accB/lB; packed 16B stores (8 contiguous dv)
  const float invF = 1.0f / lF, invB = 1.0f / lB;
  const size_t rowbase = ((size_t)b * 1024 + q) * 768 + h * 64;
#pragma unroll
  for (int t = 0; t < 4; ++t) {
    union { unsigned short s[8]; uint4 u; } o;
#pragma unroll
    for (int p = 0; p < 8; ++p)
      o.s[p] = f2bf(accF[t][p] * invF + accB[t][p] * invB);
    *reinterpret_cast<uint4*>(&attB[rowbase + t * 16 + 8 * hi]) = o.u;
  }
}

// ---------------------------------------------------------------- layernorm
__global__ __launch_bounds__(256)
void layernorm_kernel(const float* __restrict__ x, const float* __restrict__ gamma,
                      const float* __restrict__ beta, float* __restrict__ out) {
  int row = blockIdx.x * 8 + (threadIdx.x >> 5);
  int lane = threadIdx.x & 31;
  const float* xr = x + (size_t)row * 768;
  float vals[24], s = 0.f, ss = 0.f;
#pragma unroll
  for (int i = 0; i < 24; ++i) {
    float v = xr[i * 32 + lane];
    vals[i] = v; s += v; ss += v * v;
  }
#pragma unroll
  for (int m = 1; m < 32; m <<= 1) { s += __shfl_xor(s, m, 32); ss += __shfl_xor(ss, m, 32); }
  float mu = s * (1.0f / 768.0f);
  float var = ss * (1.0f / 768.0f) - mu * mu;
  float inv = rsqrtf(var + 1e-6f);
  float* orow = out + (size_t)row * 768;
#pragma unroll
  for (int i = 0; i < 24; ++i) {
    int c = i * 32 + lane;
    orow[c] = (vals[i] - mu) * inv * gamma[c] + beta[c];
  }
}

// ---------------------------------------------------------------- launch
extern "C" void kernel_launch(void* const* d_in, const int* in_sizes, int n_in,
                              void* d_out, int out_size, void* d_ws, size_t ws_size,
                              hipStream_t stream) {
  const float* q = (const float*)d_in[0];
  const float* k = (const float*)d_in[1];
  const float* v = (const float*)d_in[2];
  const unsigned char* mask = (const unsigned char*)d_in[3];  // jnp bool = 1 byte
  const float* Wq = (const float*)d_in[4];
  const float* bq = (const float*)d_in[5];
  const float* Wk = (const float*)d_in[6];
  const float* bk = (const float*)d_in[7];
  const float* Wv = (const float*)d_in[8];
  const float* bv = (const float*)d_in[9];
  const float* Wo = (const float*)d_in[10];
  const float* bo = (const float*)d_in[11];
  const float* gamma = (const float*)d_in[12];
  const float* beta = (const float*)d_in[13];

  const size_t M = 8192, D = 768;
  const size_t szAct = M * D * sizeof(unsigned short);   // 12.58 MB
  const size_t szW   = D * D * sizeof(unsigned short);   // 1.18 MB
  char* ws = (char*)d_ws;
  unsigned short* qb  = (unsigned short*)(ws + 0 * szAct);
  unsigned short* kb  = (unsigned short*)(ws + 1 * szAct);
  unsigned short* vb  = (unsigned short*)(ws + 2 * szAct);
  unsigned short* Qh  = (unsigned short*)(ws + 3 * szAct);
  unsigned short* Kh  = (unsigned short*)(ws + 4 * szAct);
  unsigned short* Vh  = (unsigned short*)(ws + 5 * szAct);
  unsigned short* atb = (unsigned short*)(ws + 6 * szAct);
  unsigned short* WqT = (unsigned short*)(ws + 7 * szAct + 0 * szW);
  unsigned short* WkT = (unsigned short*)(ws + 7 * szAct + 1 * szW);
  unsigned short* WvT = (unsigned short*)(ws + 7 * szAct + 2 * szW);
  unsigned short* WoT = (unsigned short*)(ws + 7 * szAct + 3 * szW);
  float* xbuf = (float*)(ws + 7 * szAct + 4 * szW);      // 25.17 MB f32

  const int nAct = (int)(M * D);
  // stage 0: precision conversion / weight transposition
  cvt_bf16_kernel<<<nAct / (256 * 4), 256, 0, stream>>>(q, qb, nAct);
  cvt_bf16_kernel<<<nAct / (256 * 4), 256, 0, stream>>>(k, kb, nAct);
  cvt_bf16_kernel<<<nAct / (256 * 4), 256, 0, stream>>>(v, vb, nAct);
  dim3 gt(24, 24);
  wtrans_bf16_kernel<<<gt, 256, 0, stream>>>(Wq, WqT);
  wtrans_bf16_kernel<<<gt, 256, 0, stream>>>(Wk, WkT);
  wtrans_bf16_kernel<<<gt, 256, 0, stream>>>(Wv, WvT);
  wtrans_bf16_kernel<<<gt, 256, 0, stream>>>(Wo, WoT);

  // stage 1: QKV projections (Q pre-scaled by 1/sqrt(64))
  dim3 gg(64, 6);
  gemm_bf16_kernel<0><<<gg, 256, 0, stream>>>(qb, WqT, bq, nullptr, Qh, 0.125f);
  gemm_bf16_kernel<0><<<gg, 256, 0, stream>>>(kb, WkT, bk, nullptr, Kh, 1.0f);
  gemm_bf16_kernel<0><<<gg, 256, 0, stream>>>(vb, WvT, bv, nullptr, Vh, 1.0f);

  // stage 2: bidirectional attention (transposed flash style)
  dim3 ga(16, 12, 8);
  attn_kernel<<<ga, 128, 0, stream>>>(Qh, Kh, Vh, mask, atb);

  // stage 3: output projection + residual
  gemm_bf16_kernel<1><<<gg, 256, 0, stream>>>(atb, WoT, bo, q, xbuf, 1.0f);

  // stage 4: layernorm -> d_out
  layernorm_kernel<<<1024, 256, 0, stream>>>(xbuf, gamma, beta, (float*)d_out);
}
